// TuckerExpert_72370198937818
// MI455X (gfx1250) — matrix operand order repaired and checked by reading, not compiled, not run on hardware
//
#include <hip/hip_runtime.h>
#include <hip/hip_bf16.h>

// ---------------------------------------------------------------------------
// Tucker-style MoE low-rank pipeline for MI455X (gfx1250, wave32, WMMA).
//   h   = x @ u_in                     (16384x4096 @ 4096x128)   -> WMMA f16
//   h   = einsum('tr,trs->ts', h, core[idx])  (per-token 128x128) -> VALU/L2
//   out = h @ u_out                    (16384x128 @ 128x4096)    -> WMMA f16
// HBM floor ~540MB => ~23us @ 23.3 TB/s. 64-row tiles give 256/8192 blocks
// (2+ waves/SIMD) so HBM latency is covered by co-resident waves on top of
// the double-buffered LDS pipeline.
// ---------------------------------------------------------------------------

typedef __attribute__((ext_vector_type(16))) _Float16 v16h;
typedef __attribute__((ext_vector_type(8)))  _Float16 v8h;
typedef __attribute__((ext_vector_type(4)))  _Float16 v4h;
typedef __attribute__((ext_vector_type(8)))  float    v8f;

#define TOKENS   16384
#define N_WAVES  8          // 256 threads
#define TILE_M   64         // rows per workgroup
#define TILE_N   128        // cols per workgroup
#define KC       32         // K chunk (matches 16x16x32 WMMA)

// C[M,N] = A[M,K] (f32 row major) * B[K,N] (f32 row major), K/N compile-time.
// Workgroup: 64x128 tile; 8 waves as 4 row-strips x 2 col-halves;
// each wave: 16 rows x 64 cols = 4 WMMA accumulators.
template <int K, int N>
__global__ __launch_bounds__(256)
void wmma_gemm_f16(const float* __restrict__ A, const float* __restrict__ B,
                   float* __restrict__ C) {
    __shared__ __align__(32) _Float16 a_sh[2][TILE_M * KC];   // [row][k]      2x4KB
    __shared__ __align__(32) _Float16 b_sh[2][TILE_N * KC];   // [n][k] transp 2x8KB

    const int tid     = threadIdx.x;
    const int wave    = tid >> 5;
    const int lane    = tid & 31;
    const int lanelo  = lane & 15;
    const int half    = lane >> 4;          // K-half selector per WMMA layout
    const int rowBase = blockIdx.x * TILE_M;
    const int colBase = blockIdx.y * TILE_N;
    const int waveRow = (wave >> 1) * 16;   // 4 row strips
    const int waveCol = (wave & 1) * 64;    // 2 col halves

    const float* Ab = A + (size_t)rowBase * K;
    // B staging: thread owns a 4(k) x 4(n) sub-block of the 32x128 chunk.
    const int kb = (tid >> 5) * 4;          // 0,4,..,28
    const int nb = (tid & 31) * 4;          // 0,4,..,124

    v8f acc[4];
#pragma unroll
    for (int nt = 0; nt < 4; ++nt)
#pragma unroll
        for (int i = 0; i < 8; ++i) acc[nt][i] = 0.0f;

    float4 fa[2], fb[4];

    auto load_global = [&](int k0) {
#pragma unroll
        for (int i = 0; i < 2; ++i) {       // A: 512 float4 = 64x32 f32
            int idx4 = tid + i * 256;
            int row  = idx4 >> 3;
            int kk   = (idx4 & 7) << 2;
            fa[i] = *(const float4*)(Ab + (size_t)row * K + k0 + kk);
        }
#pragma unroll
        for (int j = 0; j < 4; ++j)         // B rows k0+kb+j, cols nb..nb+3
            fb[j] = *(const float4*)(B + (size_t)(k0 + kb + j) * N + colBase + nb);
    };

    auto store_lds = [&](int buf) {
#pragma unroll
        for (int i = 0; i < 2; ++i) {       // A: packed 4-half (b64) stores
            int idx4 = tid + i * 256;
            int row  = idx4 >> 3;
            int kk   = (idx4 & 7) << 2;
            v4h p;
            p[0] = (_Float16)fa[i].x; p[1] = (_Float16)fa[i].y;
            p[2] = (_Float16)fa[i].z; p[3] = (_Float16)fa[i].w;
            *(v4h*)(&a_sh[buf][row * KC + kk]) = p;
        }
#pragma unroll
        for (int i = 0; i < 4; ++i) {       // B: transpose 4x4 in regs -> b64 stores
            const float* f0 = &fb[0].x;
            const float* f1 = &fb[1].x;
            const float* f2 = &fb[2].x;
            const float* f3 = &fb[3].x;
            v4h p;
            p[0] = (_Float16)f0[i]; p[1] = (_Float16)f1[i];
            p[2] = (_Float16)f2[i]; p[3] = (_Float16)f3[i];
            *(v4h*)(&b_sh[buf][(nb + i) * KC + kb]) = p;
        }
    };

    auto compute = [&](int buf) {
        // A fragment: lane row = waveRow+lanelo; K contiguous to match B:
        //   VGPR0-3 -> K = half*16 + {0..7}, VGPR4-7 -> K = half*16 + 8 + {0..7}
        const int arow = waveRow + lanelo;
        v8h alo = *(const v8h*)(&a_sh[buf][arow * KC + half * 16]);
        v8h ahi = *(const v8h*)(&a_sh[buf][arow * KC + half * 16 + 8]);
        v16h afrag;
#pragma unroll
        for (int i = 0; i < 8; ++i) { afrag[i] = alo[i]; afrag[8 + i] = ahi[i]; }
#pragma unroll
        for (int nt = 0; nt < 4; ++nt) {    // B: lanes 0-15 K=0..15, 16-31 K=16..31
            const int n = waveCol + nt * 16 + lanelo;
            v16h bfrag = *(const v16h*)(&b_sh[buf][n * KC + half * 16]);
            acc[nt] = __builtin_amdgcn_wmma_f32_16x16x32_f16(
                false, afrag, false, bfrag, (short)0, acc[nt], false, false);
        }
    };

    // ---- software pipeline: global loads of chunk i+1 overlap WMMA of chunk i
    constexpr int NCHUNK = K / KC;
    load_global(0);
    store_lds(0);
    __syncthreads();
    int buf = 0;
    for (int kc = 0; kc < NCHUNK; ++kc) {
        if (kc + 1 < NCHUNK) load_global((kc + 1) * KC);   // issue, don't wait
        compute(buf);                                       // WMMA from current buf
        if (kc + 1 < NCHUNK) store_lds(buf ^ 1);            // fill other buf
        __syncthreads();
        buf ^= 1;
    }

    // ---- store D: VGPR v -> M = v + 8*half, N = lanelo ----
#pragma unroll
    for (int nt = 0; nt < 4; ++nt) {
        const int col = colBase + waveCol + nt * 16 + lanelo;
#pragma unroll
        for (int v = 0; v < 8; ++v) {
            const int row = rowBase + waveRow + v + 8 * half;
            C[(size_t)row * N + col] = acc[nt][v];
        }
    }
}

// Per-token expert core contraction, in place on h:
//   h[t,:] <- sum_r h[t,r] * core[idx[t], r, :]     (128x128 per token)
// One wave per token; lane owns 4 output cols; core rows stream from L2.
__global__ __launch_bounds__(256)
void expert_mix(float* __restrict__ h, const int* __restrict__ idx,
                const float* __restrict__ core) {
    const int wave = threadIdx.x >> 5;
    const int lane = threadIdx.x & 31;
    const int t    = blockIdx.x * N_WAVES + wave;

    float4* hrow = (float4*)(h + (size_t)t * 128);
    const float4 hv = hrow[lane];                 // h[t, lane*4 .. lane*4+3]
    const int e = idx[t];
    const float4* c4 = (const float4*)(core + (size_t)e * 128 * 128);

    float4 acc = make_float4(0.f, 0.f, 0.f, 0.f);
    for (int rb = 0; rb < 32; ++rb) {             // source lane rb owns r = rb*4+j
        const float a0 = __shfl(hv.x, rb, 32);
        const float a1 = __shfl(hv.y, rb, 32);
        const float a2 = __shfl(hv.z, rb, 32);
        const float a3 = __shfl(hv.w, rb, 32);
        const float4 c0 = c4[(rb * 4 + 0) * 32 + lane];
        const float4 c1 = c4[(rb * 4 + 1) * 32 + lane];
        const float4 c2 = c4[(rb * 4 + 2) * 32 + lane];
        const float4 c3 = c4[(rb * 4 + 3) * 32 + lane];
        acc.x += a0 * c0.x + a1 * c1.x + a2 * c2.x + a3 * c3.x;
        acc.y += a0 * c0.y + a1 * c1.y + a2 * c2.y + a3 * c3.y;
        acc.z += a0 * c0.z + a1 * c1.z + a2 * c2.z + a3 * c3.z;
        acc.w += a0 * c0.w + a1 * c1.w + a2 * c2.w + a3 * c3.w;
    }
    hrow[lane] = acc;   // safe in-place: all reads of h[t] completed above
}

extern "C" void kernel_launch(void* const* d_in, const int* in_sizes, int n_in,
                              void* d_out, int out_size, void* d_ws, size_t ws_size,
                              hipStream_t stream) {
    const float* x     = (const float*)d_in[0];   // [16384, 4096]
    const int*   eidx  = (const int*)  d_in[1];   // [16384]
    const float* u_in  = (const float*)d_in[2];   // [4096, 128]
    const float* core  = (const float*)d_in[3];   // [64, 128, 128]
    const float* u_out = (const float*)d_in[4];   // [128, 4096]
    float* out = (float*)d_out;                   // [16384, 4096]
    float* h   = (float*)d_ws;                    // [16384, 128] = 8 MB scratch

    // h = x @ u_in   (K=4096, N=128) : 256 row-tiles x 1 col-tile
    wmma_gemm_f16<4096, 128><<<dim3(TOKENS / TILE_M, 1), 256, 0, stream>>>(x, u_in, h);

    // h = per-token expert contraction (in place)
    expert_mix<<<TOKENS / N_WAVES, 256, 0, stream>>>(h, eidx, core);

    // out = h @ u_out  (K=128, N=4096) : 256 x 32 tiles
    wmma_gemm_f16<128, 4096><<<dim3(TOKENS / TILE_M, 4096 / TILE_N), 256, 0, stream>>>(h, u_out, out);
}
